// GNN_89386859365067
// MI455X (gfx1250) — matrix operand order repaired and checked by reading, not compiled, not run on hardware
//
#include <hip/hip_runtime.h>

typedef __attribute__((ext_vector_type(16))) _Float16 v16h;
typedef __attribute__((ext_vector_type(4)))  _Float16 v4h;
typedef __attribute__((ext_vector_type(8)))  float    v8f;
typedef _Float16 half_t;

#define DD 128
constexpr int BSTRIDE = 136;   // padded bounce row (f16 elems) to dodge bank conflicts

// ---------------------------------------------------------------------------
// WMMA fragment helpers (layouts per CDNA5 ISA 7.12.2, wave32)
// ---------------------------------------------------------------------------

// Weight LDS layout: Wl[((k>>4)*128 + n)*16 + (k&15)]  -> B-fragment element j
// holds K = k0 + hf*8 + j, element 8+j holds K = k0 + hf*8 + 16 + j.  khi = k0/16 + hf.
__device__ __forceinline__ v16h load_bfrag(const half_t* Wl, int khi, int n) {
  const half_t* bp = Wl + ((((size_t)(khi & ~1) << 7) + (size_t)n) << 4) + (khi & 1) * 8;
  v16h b;
#pragma unroll
  for (int j = 0; j < 8; ++j) { b[j] = bp[j]; b[8 + j] = bp[2048 + j]; }
  return b;
}

// A fragment from a row-major f16 row: elements 0..7 hold K = k0 + hf*8 + j,
// elements 8..15 hold K = k0 + hf*8 + 16 + j.
__device__ __forceinline__ v16h load_afrag_row(const half_t* row, int k0, int hf) {
  const half_t* ap = row + k0 + hf * 8;
  v16h a;
#pragma unroll
  for (int j = 0; j < 8; ++j) { a[j] = ap[j]; a[8 + j] = ap[16 + j]; }
  return a;
}

__device__ __forceinline__ v16h zero_afrag() {
  v16h a;
#pragma unroll
  for (int j = 0; j < 16; ++j) a[j] = (half_t)0.f;
  return a;
}

__device__ __forceinline__ v8f wmma32(v16h a, v16h b, v8f c) {
  return __builtin_amdgcn_wmma_f32_16x16x32_f16(false, a, false, b, (short)0, c,
                                                false, false);
}

// Compute one 16x128 layer (KS k-steps of 32), LeakyReLU(0.05), write to the
// wave-private bounce tile.
template<int KS>
__device__ __forceinline__ void layer_to_bounce(const v16h* a, const half_t* Wl,
                                                const float* bias, half_t* bnc,
                                                int ln16, int hf) {
#pragma unroll
  for (int nt = 0; nt < 8; ++nt) {
    int n = nt * 16 + ln16;
    float bv = bias[n];
    v8f c = {bv, bv, bv, bv, bv, bv, bv, bv};
#pragma unroll
    for (int ks = 0; ks < KS; ++ks) {
      v16h b = load_bfrag(Wl, ks * 2 + hf, n);
      c = wmma32(a[ks], b, c);
    }
#pragma unroll
    for (int r = 0; r < 8; ++r) {
      float v = c[r];
      v = (v > 0.f) ? v : 0.05f * v;
      bnc[(r + hf * 8) * BSTRIDE + n] = (half_t)v;
    }
  }
}

__device__ __forceinline__ void load_a_bounce(v16h* a, const half_t* bnc,
                                              int ln16, int hf) {
#pragma unroll
  for (int ks = 0; ks < 4; ++ks)
    a[ks] = load_afrag_row(bnc + ln16 * BSTRIDE, ks * 32, hf);
}

// ---------------------------------------------------------------------------
// Fused MLP3 + LayerNorm (+ optional scatter-add epilogue).
//   K1S: k-steps of layer 1 (1 => padded-to-32 input, 4 => f16 [rows,128]).
//   IN16: layer-1 A comes from f16 [rows, K1S*32]; else f32 [rows, in_dim].
//   SCATTER: epilogue atomically adds rows into out[sidx[row]*128 + n].
// ---------------------------------------------------------------------------
template<int K1S, bool IN16, bool SCATTER>
__global__ __launch_bounds__(256)
void fused_mlp_ln(const float* __restrict__ Xf32, const half_t* __restrict__ Xf16,
                  int in_dim, int rows,
                  const float* __restrict__ W1, const float* __restrict__ b1,
                  const float* __restrict__ W2, const float* __restrict__ b2,
                  const float* __restrict__ W3, const float* __restrict__ b3,
                  const float* __restrict__ lng, const float* __restrict__ lnb,
                  float* __restrict__ out, const int* __restrict__ sidx) {
  extern __shared__ half_t smem[];
  const int K1 = K1S * 32;
  half_t* W1l = smem;                 // K1 x 128 (fragment layout)
  half_t* W2l = W1l + (size_t)K1 * DD;
  half_t* W3l = W2l + (size_t)DD * DD;
  half_t* bounce = W3l + (size_t)DD * DD;  // 8 waves * 16 * BSTRIDE

  const int tid = threadIdx.x;
  const int lane = tid & 31, wave = tid >> 5;
  const int hf = lane >> 4, ln16 = lane & 15;

  // cooperative weight conversion f32 -> f16 fragment layout
  for (int idx = tid; idx < K1 * DD; idx += 256) {
    int k = idx >> 7, n = idx & (DD - 1);
    float w = (k < in_dim) ? W1[k * DD + n] : 0.f;
    W1l[((((k >> 4) << 7) + n) << 4) + (k & 15)] = (half_t)w;
  }
  for (int idx = tid; idx < DD * DD; idx += 256) {
    int k = idx >> 7, n = idx & (DD - 1);
    W2l[((((k >> 4) << 7) + n) << 4) + (k & 15)] = (half_t)W2[k * DD + n];
    W3l[((((k >> 4) << 7) + n) << 4) + (k & 15)] = (half_t)W3[k * DD + n];
  }
  __syncthreads();

  const int row0 = blockIdx.x * 128 + wave * 16;
  half_t* bnc = bounce + (size_t)wave * 16 * BSTRIDE;
  const int m = row0 + ln16;

  // ---- layer 1 ----
  v16h a[4];
  if constexpr (IN16) {
#pragma unroll
    for (int ks = 0; ks < K1S; ++ks)
      a[ks] = (m < rows) ? load_afrag_row(Xf16 + (size_t)m * K1, ks * 32, hf)
                         : zero_afrag();
  } else {
    v16h av = zero_afrag();
    if (m < rows) {
      const float* xr = Xf32 + (size_t)m * in_dim;
#pragma unroll
      for (int j = 0; j < 8; ++j) {
        int k = hf * 8 + j;
        av[j] = (k < in_dim) ? (half_t)xr[k] : (half_t)0.f;
        int k2 = k + 16;
        av[8 + j] = (k2 < in_dim) ? (half_t)xr[k2] : (half_t)0.f;
      }
    }
    a[0] = av;
  }
  layer_to_bounce<K1S>(a, W1l, b1, bnc, ln16, hf);

  // ---- layer 2 ----
  load_a_bounce(a, bnc, ln16, hf);
  layer_to_bounce<4>(a, W2l, b2, bnc, ln16, hf);

  // ---- layer 3 (keep accumulators for LayerNorm) ----
  load_a_bounce(a, bnc, ln16, hf);
  v8f acc3[8];
#pragma unroll
  for (int nt = 0; nt < 8; ++nt) {
    int n = nt * 16 + ln16;
    float bv = b3[n];
    v8f c = {bv, bv, bv, bv, bv, bv, bv, bv};
#pragma unroll
    for (int ks = 0; ks < 4; ++ks) {
      v16h b = load_bfrag(W3l, ks * 2 + hf, n);
      c = wmma32(a[ks], b, c);
    }
    acc3[nt] = c;
  }

  float gv[8], bb[8];
#pragma unroll
  for (int nt = 0; nt < 8; ++nt) {
    int n = nt * 16 + ln16;
    gv[nt] = lng[n];
    bb[nt] = lnb[n];
  }

  // ---- LayerNorm per output row; row mr = row0 + hf*8 + r lives on the 16
  // lanes of this half-wave, 8 n-tiles per lane ----
#pragma unroll
  for (int r = 0; r < 8; ++r) {
    float s1 = 0.f, s2 = 0.f;
#pragma unroll
    for (int nt = 0; nt < 8; ++nt) {
      float v = acc3[nt][r];
      s1 += v; s2 += v * v;
    }
#pragma unroll
    for (int off = 1; off < 16; off <<= 1) {
      s1 += __shfl_xor(s1, off, 32);
      s2 += __shfl_xor(s2, off, 32);
    }
    float mean = s1 * (1.f / 128.f);
    float var = s2 * (1.f / 128.f) - mean * mean;
    float rstd = rsqrtf(var + 1e-5f);
    int mr = row0 + hf * 8 + r;
    if (mr < rows) {
      if constexpr (SCATTER) {
        float* pa = out + (size_t)sidx[mr] * DD + ln16;
#pragma unroll
        for (int nt = 0; nt < 8; ++nt)
          atomicAdd(pa + nt * 16, (acc3[nt][r] - mean) * rstd * gv[nt] + bb[nt]);
      } else {
        float* po = out + (size_t)mr * DD + ln16;
#pragma unroll
        for (int nt = 0; nt < 8; ++nt)
          po[nt * 16] = (acc3[nt][r] - mean) * rstd * gv[nt] + bb[nt];
      }
    }
  }
}

// ---------------------------------------------------------------------------
// Plain WMMA GEMM: out[rows,128] = A_f16[rows,128] @ W_f32[128,128]
// ---------------------------------------------------------------------------
__global__ __launch_bounds__(256)
void gemm128_wmma(const half_t* __restrict__ A, const float* __restrict__ Wg,
                  float* __restrict__ out, int rows) {
  __shared__ half_t Wl[DD * DD];
  const int tid = threadIdx.x;
  const int lane = tid & 31, wave = tid >> 5;
  const int hf = lane >> 4, ln16 = lane & 15;
  for (int idx = tid; idx < DD * DD; idx += 256) {
    int k = idx >> 7, n = idx & (DD - 1);
    Wl[((((k >> 4) << 7) + n) << 4) + (k & 15)] = (half_t)Wg[k * DD + n];
  }
  __syncthreads();

  const int row0 = blockIdx.x * 128 + wave * 16;
  const int m = row0 + ln16;
  v16h a[4];
#pragma unroll
  for (int ks = 0; ks < 4; ++ks)
    a[ks] = (m < rows) ? load_afrag_row(A + (size_t)m * DD, ks * 32, hf)
                       : zero_afrag();

  const int mbase = row0 + hf * 8;
  const bool full = (mbase + 8 <= rows);
#pragma unroll
  for (int nt = 0; nt < 8; ++nt) {
    int n = nt * 16 + ln16;
    v8f c = {0.f, 0.f, 0.f, 0.f, 0.f, 0.f, 0.f, 0.f};
#pragma unroll
    for (int ks = 0; ks < 4; ++ks) {
      v16h b = load_bfrag(Wl, ks * 2 + hf, n);
      c = wmma32(a[ks], b, c);
    }
    float* po = out + (size_t)mbase * DD + n;
    if (full) {
#pragma unroll
      for (int r = 0; r < 8; ++r) po[r * DD] = c[r];   // branchless fast path
    } else {
#pragma unroll
      for (int r = 0; r < 8; ++r)
        if (mbase + r < rows) po[r * DD] = c[r];
    }
  }
}

// ---------------------------------------------------------------------------
// Small utility kernels
// ---------------------------------------------------------------------------
__global__ void k_zero(float* p, long long n) {
  long long i = (long long)blockIdx.x * 256 + threadIdx.x;
  if (i < n) p[i] = 0.f;
}

__global__ void k_zero4(float4* p, long long nq) {
  long long i = (long long)blockIdx.x * 256 + threadIdx.x;
  if (i < nq) p[i] = make_float4(0.f, 0.f, 0.f, 0.f);
}

__global__ void k_count(const int* __restrict__ idx, float* cnt, int E) {
  int e = blockIdx.x * 256 + threadIdx.x;
  if (e < E) atomicAdd(&cnt[idx[e]], 1.f);
}

__global__ void k_dis(const float* __restrict__ degsum, float* dis, int N) {
  int i = blockIdx.x * 256 + threadIdx.x;
  if (i < N) dis[i] = rsqrtf(degsum[i] + 1.f);
}

// hn += eacc / max(cnt,1); refresh hn16.  One float4 per thread.
__global__ void k_apply_mean(float* __restrict__ hn, const float* __restrict__ eacc,
                             const float* __restrict__ cnt, half_t* __restrict__ hn16,
                             int N) {
  long long q = (long long)blockIdx.x * 256 + threadIdx.x;
  if (q < (long long)N * (DD / 4)) {
    int i = (int)(q >> 5);                      // 32 quads per row
    float4 hv = ((const float4*)hn)[q];
    float4 ev = ((const float4*)eacc)[q];
    float inv = 1.f / fmaxf(cnt[i], 1.f);
    hv.x += ev.x * inv; hv.y += ev.y * inv; hv.z += ev.z * inv; hv.w += ev.w * inv;
    ((float4*)hn)[q] = hv;
    v4h o; o[0] = (half_t)hv.x; o[1] = (half_t)hv.y; o[2] = (half_t)hv.z; o[3] = (half_t)hv.w;
    ((v4h*)hn16)[q] = o;
  }
}

// edge message: agg[dst] += h[src] * dis[src]*dis[dst].
// One edge per wave: uniform indices go scalar, lanes move float4 each.
__global__ __launch_bounds__(256)
void k_edge_msg(const float* __restrict__ h, const int* __restrict__ src,
                const int* __restrict__ dst, const float* __restrict__ dis,
                float* __restrict__ agg, int E) {
  int wave = threadIdx.x >> 5, lane = threadIdx.x & 31;
  int e = blockIdx.x * 8 + wave;
  if (e >= E) return;
  int s = __builtin_amdgcn_readfirstlane(src[e]);
  int t = __builtin_amdgcn_readfirstlane(dst[e]);
  float c = dis[s] * dis[t];
  float4 v = ((const float4*)(h + (size_t)s * DD))[lane];
  float* ap = agg + (size_t)t * DD + lane * 4;
  atomicAdd(ap + 0, v.x * c);
  atomicAdd(ap + 1, v.y * c);
  atomicAdd(ap + 2, v.z * c);
  atomicAdd(ap + 3, v.w * c);
}

// conv1 epilogue: y = leaky(agg + h*dis^2 + b); accumulate BN sums.
__global__ __launch_bounds__(256)
void k_combine1(const float* __restrict__ agg, const float* __restrict__ h,
                const float* __restrict__ dis, const float* __restrict__ bias,
                float* __restrict__ y, float* __restrict__ bnstats, int N) {
  int d = threadIdx.x & (DD - 1);
  int grp = threadIdx.x >> 7;
  int i0 = blockIdx.x * 64 + grp * 32;
  float b = bias[d];
  float s1 = 0.f, s2 = 0.f;
  for (int r = 0; r < 32; ++r) {
    int i = i0 + r;
    if (i < N) {
      float di = dis[i];
      size_t o = (size_t)i * DD + d;
      float v = agg[o] + h[o] * di * di + b;
      v = (v > 0.f) ? v : 0.05f * v;
      y[o] = v;
      s1 += v;
      s2 += v * v;
    }
  }
  atomicAdd(&bnstats[d], s1);
  atomicAdd(&bnstats[DD + d], s2);
}

__global__ void k_bn_finalize(const float* __restrict__ bnstats,
                              const float* __restrict__ g,
                              const float* __restrict__ b, float* __restrict__ prm,
                              int N) {
  int d = threadIdx.x;  // 128 threads
  float inv = 1.f / (float)N;
  float m = bnstats[d] * inv;
  float var = bnstats[DD + d] * inv - m * m;
  float sc = rsqrtf(var + 1e-5f) * g[d];
  prm[d] = sc;
  prm[DD + d] = b[d] - m * sc;
}

__global__ void k_bn_apply(const float* __restrict__ y, const float* __restrict__ prm,
                           half_t* __restrict__ bn16, int N) {
  long long q = (long long)blockIdx.x * 256 + threadIdx.x;
  if (q < (long long)N * (DD / 4)) {
    int d4 = (int)(q & 31);
    float4 yv = ((const float4*)y)[q];
    float4 sc = ((const float4*)prm)[d4];
    float4 sh = ((const float4*)(prm + DD))[d4];
    v4h o;
    o[0] = (half_t)(yv.x * sc.x + sh.x);
    o[1] = (half_t)(yv.y * sc.y + sh.y);
    o[2] = (half_t)(yv.z * sc.z + sh.z);
    o[3] = (half_t)(yv.w * sc.w + sh.w);
    ((v4h*)bn16)[q] = o;
  }
}

// conv2 epilogue + residual: hn += agg + h*dis^2 + b ; refresh hn16
__global__ void k_combine2(const float* __restrict__ agg, const float* __restrict__ h,
                           const float* __restrict__ dis, const float* __restrict__ bias,
                           float* __restrict__ hn, half_t* __restrict__ hn16, int N) {
  long long q = (long long)blockIdx.x * 256 + threadIdx.x;
  if (q < (long long)N * (DD / 4)) {
    int i = (int)(q >> 5);
    int d4 = (int)(q & 31);
    float di = dis[i];
    float d2 = di * di;
    float4 hv = ((const float4*)hn)[q];
    float4 av = ((const float4*)agg)[q];
    float4 xv = ((const float4*)h)[q];
    float4 bv = ((const float4*)bias)[d4];
    hv.x += av.x + xv.x * d2 + bv.x;
    hv.y += av.y + xv.y * d2 + bv.y;
    hv.z += av.z + xv.z * d2 + bv.z;
    hv.w += av.w + xv.w * d2 + bv.w;
    ((float4*)hn)[q] = hv;
    v4h o; o[0] = (half_t)hv.x; o[1] = (half_t)hv.y; o[2] = (half_t)hv.z; o[3] = (half_t)hv.w;
    ((v4h*)hn16)[q] = o;
  }
}

// ---------------------------------------------------------------------------
extern "C" void kernel_launch(void* const* d_in, const int* in_sizes, int n_in,
                              void* d_out, int out_size, void* d_ws, size_t ws_size,
                              hipStream_t stream) {
  const float* node_feat = (const float*)d_in[0];
  const float* edge_feat = (const float*)d_in[1];
  const int*   edge_index = (const int*)d_in[2];
  const float* enW1 = (const float*)d_in[3];  const float* enb1 = (const float*)d_in[4];
  const float* enW2 = (const float*)d_in[5];  const float* enb2 = (const float*)d_in[6];
  const float* enW3 = (const float*)d_in[7];  const float* enb3 = (const float*)d_in[8];
  const float* enlg = (const float*)d_in[9];  const float* enlb = (const float*)d_in[10];
  const float* eeW1 = (const float*)d_in[11]; const float* eeb1 = (const float*)d_in[12];
  const float* eeW2 = (const float*)d_in[13]; const float* eeb2 = (const float*)d_in[14];
  const float* eeW3 = (const float*)d_in[15]; const float* eeb3 = (const float*)d_in[16];
  const float* eelg = (const float*)d_in[17]; const float* eelb = (const float*)d_in[18];
  const float* procW = (const float*)d_in[19];
  const float* procB = (const float*)d_in[20];
  const float* bnG = (const float*)d_in[21];
  const float* bnB = (const float*)d_in[22];
  const float* dW1 = (const float*)d_in[23]; const float* db1 = (const float*)d_in[24];
  const float* dW2 = (const float*)d_in[25]; const float* db2 = (const float*)d_in[26];
  const float* dW3 = (const float*)d_in[27]; const float* db3 = (const float*)d_in[28];
  const float* dlg = (const float*)d_in[29]; const float* dlb = (const float*)d_in[30];

  const int N = in_sizes[0] / 19;
  const int E = in_sizes[1] / 4;
  const int* src = edge_index;
  const int* dst = edge_index + E;
  float* out = (float*)d_out;

  // workspace carve-up (256B aligned)
  char* p = (char*)d_ws;
  auto alloc = [&](size_t bytes) -> char* {
    char* r = p;
    p += (bytes + 255) & ~(size_t)255;
    return r;
  };
  const size_t ND = (size_t)N * DD;
  float*  hn     = (float*) alloc(ND * 4);
  half_t* hn16   = (half_t*)alloc(ND * 2);
  float*  eacc   = (float*) alloc(ND * 4);
  float*  hbuf   = (float*) alloc(ND * 4);
  float*  ybuf   = (float*) alloc(ND * 4);
  float*  agg    = (float*) alloc(ND * 4);
  half_t* bn16   = (half_t*)alloc(ND * 2);
  float*  cnt    = (float*) alloc((size_t)N * 4);
  float*  degsum = (float*) alloc((size_t)N * 4);
  float*  dis    = (float*) alloc((size_t)N * 4);
  float*  bnstats= (float*) alloc(256 * 4);

  const int TB = 256;
  auto cdiv = [](long long a, long long b) { return (unsigned)((a + b - 1) / b); };

  const size_t ldsEnc = (size_t)(32 * DD + 2 * DD * DD + 8 * 16 * BSTRIDE) * sizeof(half_t);
  const size_t ldsDec = (size_t)(3 * DD * DD + 8 * 16 * BSTRIDE) * sizeof(half_t);
  const unsigned gN = cdiv(N, 128), gE = cdiv(E, 128);
  const long long NQ = (long long)ND / 4;          // float4 quads
  float* bnprm = bnstats + 0;                      // finalize overwrites in place? no:
  // keep separate small buffer for bn params
  float* bnprm2 = (float*)alloc(256 * 4);

  // degree / count accumulators
  k_zero<<<cdiv(N, TB), TB, 0, stream>>>(cnt, N);
  k_zero<<<cdiv(N, TB), TB, 0, stream>>>(degsum, N);
  k_zero4<<<cdiv(NQ, TB), TB, 0, stream>>>((float4*)eacc, NQ);
  k_count<<<cdiv(E, TB), TB, 0, stream>>>(src, cnt, E);
  k_count<<<cdiv(E, TB), TB, 0, stream>>>(dst, degsum, E);
  k_dis<<<cdiv(N, TB), TB, 0, stream>>>(degsum, dis, N);

  // node encoder -> hn (f32)
  fused_mlp_ln<1, false, false><<<gN, TB, ldsEnc, stream>>>(
      node_feat, nullptr, 19, N, enW1, enb1, enW2, enb2, enW3, enb3, enlg, enlb,
      hn, nullptr);
  // edge encoder, fused scatter-add of LayerNorm output onto src nodes
  fused_mlp_ln<1, false, true><<<gE, TB, ldsEnc, stream>>>(
      edge_feat, nullptr, 4, E, eeW1, eeb1, eeW2, eeb2, eeW3, eeb3, eelg, eelb,
      eacc, src);
  // hn += eacc / max(cnt,1); produce hn16
  k_apply_mean<<<cdiv(NQ, TB), TB, 0, stream>>>(hn, eacc, cnt, hn16, N);

  // 5 processor steps
  for (int s = 0; s < 5; ++s) {
    const float* W0 = procW + (size_t)(s * 2 + 0) * DD * DD;
    const float* W1p = procW + (size_t)(s * 2 + 1) * DD * DD;
    const float* b0 = procB + (size_t)(s * 2 + 0) * DD;
    const float* b1p = procB + (size_t)(s * 2 + 1) * DD;

    // conv1: h = hn @ W0
    gemm128_wmma<<<gN, TB, 0, stream>>>(hn16, W0, hbuf, N);
    k_zero4<<<cdiv(NQ, TB), TB, 0, stream>>>((float4*)agg, NQ);
    k_edge_msg<<<cdiv(E, 8), TB, 0, stream>>>(hbuf, src, dst, dis, agg, E);
    k_zero<<<1, TB, 0, stream>>>(bnstats, 256);
    k_combine1<<<cdiv(N, 64), TB, 0, stream>>>(agg, hbuf, dis, b0, ybuf, bnstats, N);
    k_bn_finalize<<<1, DD, 0, stream>>>(bnstats, bnG + s * DD, bnB + s * DD, bnprm2, N);
    k_bn_apply<<<cdiv(NQ, TB), TB, 0, stream>>>(ybuf, bnprm2, bn16, N);

    // conv2: h2 = bn(x) @ W1p ; residual into hn
    gemm128_wmma<<<gN, TB, 0, stream>>>(bn16, W1p, hbuf, N);
    k_zero4<<<cdiv(NQ, TB), TB, 0, stream>>>((float4*)agg, NQ);
    k_edge_msg<<<cdiv(E, 8), TB, 0, stream>>>(hbuf, src, dst, dis, agg, E);
    k_combine2<<<cdiv(NQ, TB), TB, 0, stream>>>(agg, hbuf, dis, b1p, hn, hn16, N);
  }

  // decoder -> d_out
  fused_mlp_ln<4, true, false><<<gN, TB, ldsDec, stream>>>(
      nullptr, hn16, DD, N, dW1, db1, dW2, db2, dW3, db3, dlg, dlb, out, nullptr);
  (void)bnprm; (void)ws_size; (void)n_in; (void)out_size;
}
